// TransformerEncoderLayer_4810363372428
// MI455X (gfx1250) — matrix operand order repaired and checked by reading, not compiled, not run on hardware
//
#include <hip/hip_runtime.h>
#include <hip/hip_bf16.h>

typedef __attribute__((ext_vector_type(16))) _Float16 v16h;
typedef __attribute__((ext_vector_type(8)))  _Float16 v8h;
typedef __attribute__((ext_vector_type(8)))  float    v8f;
typedef __attribute__((ext_vector_type(4)))  unsigned int v4u;
typedef __attribute__((ext_vector_type(8)))  int          v8i;
typedef __attribute__((ext_vector_type(4)))  int          v4i;

#define DIMSZ 512
#define LSEQ  2048
#define BATCH 2
#define NHEAD 8
#define DHEAD 64
#define MLPD  2048
#define ROWS  (BATCH*LSEQ)   // 4096

#if defined(__has_builtin)
#  if __has_builtin(__builtin_amdgcn_tensor_load_to_lds)
#    define HAVE_TDM 1
#  else
#    define HAVE_TDM 0
#  endif
#else
#  define HAVE_TDM 0
#endif

static __device__ inline v16h cat8(v8h lo, v8h hi) {
  return __builtin_shufflevector(lo, hi, 0,1,2,3,4,5,6,7,8,9,10,11,12,13,14,15);
}

#if HAVE_TDM
// One 2D TDM tile load: tile_h rows x tile_w f16 (contiguous), global row
// stride `stride` (elements), LDS dest padded: after every 2^(pi+1) dwords
// insert (pa+1) dwords of padding.  D# layout per CDNA5 ISA ch.8.
static __device__ inline void tdm_load_2d(unsigned lds_off, const void* gaddr,
                                          int tile_w, int tile_h, int stride,
                                          int pi, int pa) {
  unsigned long long ga = (unsigned long long)(uintptr_t)gaddr;
  v4u g0;
  g0[0] = 1u;                                    // count=1, user desc
  g0[1] = lds_off;                               // lds_addr (bytes)
  g0[2] = (unsigned)(ga & 0xFFFFFFFFu);          // global_addr[31:0]
  g0[3] = (unsigned)((ga >> 32) & 0x01FFFFFFu) | (2u << 30);  // addr[56:32]|type=2
  v8i g1;
  g1[0] = (int)((1u << 16) | (1u << 20) |        // data_size=2B, pad_enable
                ((unsigned)pi << 22) | ((unsigned)pa << 25));
  g1[1] = (int)(((unsigned)tile_w & 0xFFFFu) << 16);                  // tensor_dim0 lo
  g1[2] = (int)((((unsigned)tile_w >> 16) & 0xFFFFu) |
                (((unsigned)tile_h & 0xFFFFu) << 16));                // dim0 hi|dim1 lo
  g1[3] = (int)((((unsigned)tile_h >> 16) & 0xFFFFu) |
                (((unsigned)tile_w & 0xFFFFu) << 16));                // dim1 hi|tile_dim0
  g1[4] = (int)((unsigned)tile_h & 0xFFFFu);                          // tile_dim1, tile_dim2=0
  g1[5] = (int)(unsigned)stride;                                      // dim0_stride[31:0]
  g1[6] = 0;
  g1[7] = 0;
  v4i g2 = {0, 0, 0, 0};
  v4i g3 = {0, 0, 0, 0};
#if __has_include(<hip/amd_detail/amd_gfx1250_TDM.h>)
  v8i gx = {0, 0, 0, 0, 0, 0, 0, 0};
  __builtin_amdgcn_tensor_load_to_lds(g0, g1, g2, g3, gx, 0);
#else
  __builtin_amdgcn_tensor_load_to_lds(g0, g1, g2, g3, 0);
#endif
}
static __device__ inline unsigned lds_off_of(const void* p) {
  return (unsigned)(uintptr_t)p;   // low 32 bits of generic addr = LDS offset
}
#endif

// ---------------- fp32 -> fp16 transposed weight conversion ----------------
// src [K,N] row-major (fp32)  ->  dst [N,K] row-major (f16)
__global__ void convert_transpose_kernel(const float* __restrict__ src,
                                         _Float16* __restrict__ dst, int K, int N) {
  int i = blockIdx.x * 256 + threadIdx.x;
  if (i < K * N) {
    int k = i / N, n = i - k * N;
    dst[(size_t)n * K + k] = (_Float16)src[i];
  }
}

// ---------------- LayerNorm (fp32 in -> fp16 out), one row per block -------
__global__ __launch_bounds__(128) void ln_kernel(const float* __restrict__ x,
                                                 const float* __restrict__ w,
                                                 const float* __restrict__ b,
                                                 _Float16* __restrict__ out) {
  int row = blockIdx.x;
  const float* xr = x + (size_t)row * DIMSZ;
  int tid = threadIdx.x;
  float v0[4];
  float s = 0.f, ss = 0.f;
#pragma unroll
  for (int i = 0; i < 4; ++i) {
    float t = xr[tid + i * 128];
    v0[i] = t; s += t; ss += t * t;
  }
#pragma unroll
  for (int off = 1; off < 32; off <<= 1) {
    s  += __shfl_xor(s,  off);
    ss += __shfl_xor(ss, off);
  }
  __shared__ float red[8];
  int wv = tid >> 5, ln = tid & 31;
  if (ln == 0) { red[wv] = s; red[4 + wv] = ss; }
  __syncthreads();
  if (tid == 0) {
    float S  = red[0] + red[1] + red[2] + red[3];
    float SS = red[4] + red[5] + red[6] + red[7];
    float mean = S * (1.f / DIMSZ);
    float var  = SS * (1.f / DIMSZ) - mean * mean;
    red[0] = mean;
    red[1] = rsqrtf(var + 1e-6f);
  }
  __syncthreads();
  float mean = red[0], rstd = red[1];
#pragma unroll
  for (int i = 0; i < 4; ++i) {
    int j = tid + i * 128;
    out[(size_t)row * DIMSZ + j] = (_Float16)((v0[i] - mean) * rstd * w[j] + b[j]);
  }
}

// ---------------- WMMA GEMM: C = A[M,K](f16) * BT[N,K](f16)^T + bias -------
// Block tile 128x128, 8 waves (4x2), wave tile 32x64 (2 A-frags x 4 B-frags).
// EPI 0: f16 out (bias).  EPI 1: f16 out (bias + exact GELU).
// EPI 2: f32 out (bias + residual add from `res`).
// EPI 3: qkv split: n<1024 -> f16 out [m,1536]; n>=1024 -> V^T into out2.
template <int EPI>
__global__ __launch_bounds__(256) void gemm_kernel(
    const _Float16* __restrict__ A, const _Float16* __restrict__ BT,
    const float* __restrict__ bias, const float* __restrict__ res,
    void* __restrict__ outp, _Float16* __restrict__ out2,
    int M, int N, int K) {
  __shared__ _Float16 As[2][128 * 40];   // [m][k] padded stride 40
  __shared__ _Float16 Bs[2][128 * 40];   // [n][k] padded stride 40
  int tid = threadIdx.x;
  int wv = tid >> 5, lane = tid & 31, li = lane & 15, hihalf = lane >> 4;
  int m0 = blockIdx.y * 128, n0 = blockIdx.x * 128;
  int waveM = wv >> 1, waveN = wv & 1;   // 4 x 2 wave grid
  v8f acc[2][4] = {};
  const int T = K >> 5;

  // ---- tile staging (TDM double-buffered, or cooperative-copy fallback) ----
#if HAVE_TDM
  auto stage = [&](int it, int bb) {
    if (wv == 0) {
      int k0 = it * 32;
      tdm_load_2d(lds_off_of(&As[bb][0]), A  + (size_t)m0 * K + k0, 32, 128, K, 3, 3);
      tdm_load_2d(lds_off_of(&Bs[bb][0]), BT + (size_t)n0 * K + k0, 32, 128, K, 3, 3);
    }
  };
#else
  auto stage = [&](int it, int bb) {
    int k0 = it * 32;
#pragma unroll
    for (int i = 0; i < 2; ++i) {
      int c = tid + i * 256;
      int row = c >> 2, seg = c & 3;
      *(v8h*)(&As[bb][0] + row * 40 + seg * 8) =
          *(const v8h*)(A + (size_t)(m0 + row) * K + k0 + seg * 8);
      *(v8h*)(&Bs[bb][0] + row * 40 + seg * 8) =
          *(const v8h*)(BT + (size_t)(n0 + row) * K + k0 + seg * 8);
    }
  };
#endif

  int buf = 0;
  stage(0, 0);
  for (int it = 0; it < T; ++it) {
    __syncthreads();                       // buf^1 consumers (it-1) done
    if (it + 1 < T) stage(it + 1, buf ^ 1);
#if HAVE_TDM
    if (wv == 0) {
      if (it + 1 < T) __builtin_amdgcn_s_wait_tensorcnt(2);
      else            __builtin_amdgcn_s_wait_tensorcnt(0);
    }
#endif
    __syncthreads();                       // tile `it` visible to all waves

    v16h aF[2], bF[4];
#pragma unroll
    for (int t = 0; t < 2; ++t) {
      const _Float16* ab = &As[buf][0] + (32 * waveM + 16 * t + li) * 40 + 8 * hihalf;
      aF[t] = cat8(*(const v8h*)ab, *(const v8h*)(ab + 16));
    }
#pragma unroll
    for (int u = 0; u < 4; ++u) {
      const _Float16* bb = &Bs[buf][0] + (64 * waveN + 16 * u + li) * 40 + 8 * hihalf;
      bF[u] = cat8(*(const v8h*)bb, *(const v8h*)(bb + 16));
    }
#pragma unroll
    for (int t = 0; t < 2; ++t)
#pragma unroll
      for (int u = 0; u < 4; ++u)
        acc[t][u] = __builtin_amdgcn_wmma_f32_16x16x32_f16(
            false, aF[t], false, bF[u], (short)0, acc[t][u], false, false);
    buf ^= 1;
  }

  // epilogue: C layout -> row = r + 8*hihalf, col = li
#pragma unroll
  for (int t = 0; t < 2; ++t)
#pragma unroll
    for (int u = 0; u < 4; ++u)
#pragma unroll
      for (int r = 0; r < 8; ++r) {
        int m = m0 + 32 * waveM + 16 * t + r + 8 * hihalf;
        int n = n0 + 64 * waveN + 16 * u + li;
        float v = acc[t][u][r] + bias[n];
        if constexpr (EPI == 1)
          v = 0.5f * v * (1.0f + erff(v * 0.70710678118654752f));
        if constexpr (EPI == 2) {
          ((float*)outp)[(size_t)m * N + n] = v + res[(size_t)m * N + n];
        } else if constexpr (EPI == 3) {
          if (n < 2 * DIMSZ) {             // Q,K sections: row-major qkv buffer
            ((_Float16*)outp)[(size_t)m * (3 * DIMSZ) + n] = (_Float16)v;
          } else {                         // V section: transposed [b,h,d,l]
            int hd = n - 2 * DIMSZ;
            int hh = hd >> 6, d = hd & 63;
            int bb2 = m >> 11, l = m & (LSEQ - 1);
            out2[(((size_t)bb2 * NHEAD + hh) * DHEAD + d) * LSEQ + l] = (_Float16)v;
          }
        } else {
          ((_Float16*)outp)[(size_t)m * N + n] = (_Float16)v;
        }
      }
}

// ---------------- Flash attention: block = (b, h, 64 q rows), 4 waves ------
__global__ __launch_bounds__(128) void attn_kernel(
    const _Float16* __restrict__ qkv, const _Float16* __restrict__ vT,
    _Float16* __restrict__ o_out) {
  __shared__ _Float16 Kt[2][32 * 72];       // [key][dim], stride 72
  __shared__ _Float16 Vt[2][64 * 40];       // [dim][key], stride 40
  __shared__ _Float16 Pbuf[4 * 16 * 40];    // per-wave P, [row][key], stride 40
  int tid = threadIdx.x;
  int wv = tid >> 5, lane = tid & 31, li = lane & 15, hihalf = lane >> 4;
  int b = blockIdx.z, h = blockIdx.y, qt = blockIdx.x;
  const _Float16* base = qkv + (size_t)b * LSEQ * (3 * DIMSZ);
  const _Float16* Qp = base + h * DHEAD;
  const _Float16* Kp = base + DIMSZ + h * DHEAD;
  const _Float16* Vp = vT + ((size_t)(b * NHEAD + h) * DHEAD) * LSEQ;  // [64][2048]
  int qrow = qt * 64 + wv * 16 + li;

  v16h aQ[2];
#pragma unroll
  for (int f = 0; f < 2; ++f) {
    const _Float16* qp = Qp + (size_t)qrow * (3 * DIMSZ) + f * 32 + 8 * hihalf;
    aQ[f] = cat8(*(const v8h*)qp, *(const v8h*)(qp + 16));
  }

  v8f o[4] = {};
  float mrow[8], lrow[8];
#pragma unroll
  for (int r = 0; r < 8; ++r) { mrow[r] = -1e30f; lrow[r] = 0.f; }
  _Float16* Pw = Pbuf + wv * 640;
  const int T = LSEQ / 32;

#if HAVE_TDM
  auto stage = [&](int it, int bb) {
    if (wv == 0) {
      int kb = it * 32;
      tdm_load_2d(lds_off_of(&Kt[bb][0]), Kp + (size_t)kb * (3 * DIMSZ),
                  64, 32, 3 * DIMSZ, 4, 3);
      tdm_load_2d(lds_off_of(&Vt[bb][0]), Vp + kb, 32, 64, LSEQ, 3, 3);
    }
  };
#else
  auto stage = [&](int it, int bb) {
    int kb = it * 32;
#pragma unroll
    for (int i = 0; i < 2; ++i) {
      int c = tid + i * 128;
      int key = c >> 3, seg = c & 7;
      *(v8h*)(&Kt[bb][0] + key * 72 + seg * 8) =
          *(const v8h*)(Kp + (size_t)(kb + key) * (3 * DIMSZ) + seg * 8);
    }
#pragma unroll
    for (int i = 0; i < 2; ++i) {
      int c = tid + i * 128;
      int row = c >> 2, seg = c & 3;
      *(v8h*)(&Vt[bb][0] + row * 40 + seg * 8) =
          *(const v8h*)(Vp + (size_t)row * LSEQ + kb + seg * 8);
    }
  };
#endif

  int buf = 0;
  stage(0, 0);
  for (int it = 0; it < T; ++it) {
    __syncthreads();
    if (it + 1 < T) stage(it + 1, buf ^ 1);
#if HAVE_TDM
    if (wv == 0) {
      if (it + 1 < T) __builtin_amdgcn_s_wait_tensorcnt(2);
      else            __builtin_amdgcn_s_wait_tensorcnt(0);
    }
#endif
    __syncthreads();

    // S = Q K^T : two 16x16 tiles
    v8f s[2];
#pragma unroll
    for (int t = 0; t < 2; ++t) {
      v8f z = {};
#pragma unroll
      for (int f = 0; f < 2; ++f) {
        const _Float16* kb8 = &Kt[buf][0] + (t * 16 + li) * 72 + f * 32 + 8 * hihalf;
        v16h bk = cat8(*(const v8h*)kb8, *(const v8h*)(kb8 + 16));
        z = __builtin_amdgcn_wmma_f32_16x16x32_f16(
            false, aQ[f], false, bk, (short)0, z, false, false);
      }
      s[t] = z;
    }

    // online softmax (row = r + 8*hihalf; reduce across 16-lane half)
#pragma unroll
    for (int r = 0; r < 8; ++r) {
      float s0 = s[0][r] * 0.125f, s1 = s[1][r] * 0.125f;
      float mx = fmaxf(s0, s1);
#pragma unroll
      for (int off = 1; off < 16; off <<= 1) mx = fmaxf(mx, __shfl_xor(mx, off));
      float mn = fmaxf(mrow[r], mx);
      float alpha = __expf(mrow[r] - mn);
      mrow[r] = mn;
      float p0 = __expf(s0 - mn), p1 = __expf(s1 - mn);
      float rs = p0 + p1;
#pragma unroll
      for (int off = 1; off < 16; off <<= 1) rs += __shfl_xor(rs, off);
      lrow[r] = lrow[r] * alpha + rs;
#pragma unroll
      for (int g = 0; g < 4; ++g) o[g][r] *= alpha;
      int prow = r + 8 * hihalf;
      Pw[prow * 40 + li]      = (_Float16)p0;
      Pw[prow * 40 + 16 + li] = (_Float16)p1;
    }

    // P (C-layout) -> A-layout via per-wave LDS round trip
    const _Float16* pp = Pw + li * 40 + 8 * hihalf;
    v16h aP = cat8(*(const v8h*)pp, *(const v8h*)(pp + 16));

    // O += P * V
#pragma unroll
    for (int g = 0; g < 4; ++g) {
      const _Float16* vb = &Vt[buf][0] + (g * 16 + li) * 40 + 8 * hihalf;
      v16h bv = cat8(*(const v8h*)vb, *(const v8h*)(vb + 16));
      o[g] = __builtin_amdgcn_wmma_f32_16x16x32_f16(
          false, aP, false, bv, (short)0, o[g], false, false);
    }
    buf ^= 1;
  }

#pragma unroll
  for (int g = 0; g < 4; ++g)
#pragma unroll
    for (int r = 0; r < 8; ++r) {
      int q = qt * 64 + wv * 16 + r + 8 * hihalf;
      int col = h * DHEAD + g * 16 + li;
      o_out[((size_t)b * LSEQ + q) * DIMSZ + col] = (_Float16)(o[g][r] / lrow[r]);
    }
}

// ---------------------------------------------------------------------------
extern "C" void kernel_launch(void* const* d_in, const int* in_sizes, int n_in,
                              void* d_out, int out_size, void* d_ws, size_t ws_size,
                              hipStream_t stream) {
  const float* x     = (const float*)d_in[0];
  const float* ln1_w = (const float*)d_in[1];
  const float* ln1_b = (const float*)d_in[2];
  const float* qkv_w = (const float*)d_in[3];
  const float* qkv_b = (const float*)d_in[4];
  const float* out_w = (const float*)d_in[5];
  const float* out_b = (const float*)d_in[6];
  const float* ln2_w = (const float*)d_in[7];
  const float* ln2_b = (const float*)d_in[8];
  const float* fc1_w = (const float*)d_in[9];
  const float* fc1_b = (const float*)d_in[10];
  const float* fc2_w = (const float*)d_in[11];
  const float* fc2_b = (const float*)d_in[12];

  char* ws = (char*)d_ws;
  size_t off = 0;
  auto alloc = [&](size_t bytes) -> void* {
    void* p = ws + off;
    off += (bytes + 255) & ~(size_t)255;
    return p;
  };

  _Float16* qkvWT = (_Float16*)alloc((size_t)DIMSZ * 3 * DIMSZ * 2); // [1536][512]
  _Float16* outWT = (_Float16*)alloc((size_t)DIMSZ * DIMSZ * 2);     // [512][512]
  _Float16* fc1WT = (_Float16*)alloc((size_t)DIMSZ * MLPD * 2);      // [2048][512]
  _Float16* fc2WT = (_Float16*)alloc((size_t)MLPD * DIMSZ * 2);      // [512][2048]
  _Float16* hbuf  = (_Float16*)alloc((size_t)ROWS * DIMSZ * 2);
  _Float16* qkv16 = (_Float16*)alloc((size_t)ROWS * 3 * DIMSZ * 2);
  _Float16* vT    = (_Float16*)alloc((size_t)BATCH * NHEAD * DHEAD * LSEQ * 2);
  _Float16* obuf  = (_Float16*)alloc((size_t)ROWS * DIMSZ * 2);
  float*    x1    = (float*)   alloc((size_t)ROWS * DIMSZ * 4);
  _Float16* h2    = (_Float16*)alloc((size_t)ROWS * DIMSZ * 2);
  _Float16* a16   = (_Float16*)alloc((size_t)ROWS * MLPD * 2);

  // transposed weight conversion
  int nw;
  nw = DIMSZ * 3 * DIMSZ;
  convert_transpose_kernel<<<(nw + 255) / 256, 256, 0, stream>>>(qkv_w, qkvWT, DIMSZ, 3 * DIMSZ);
  nw = DIMSZ * DIMSZ;
  convert_transpose_kernel<<<(nw + 255) / 256, 256, 0, stream>>>(out_w, outWT, DIMSZ, DIMSZ);
  nw = DIMSZ * MLPD;
  convert_transpose_kernel<<<(nw + 255) / 256, 256, 0, stream>>>(fc1_w, fc1WT, DIMSZ, MLPD);
  nw = MLPD * DIMSZ;
  convert_transpose_kernel<<<(nw + 255) / 256, 256, 0, stream>>>(fc2_w, fc2WT, MLPD, DIMSZ);

  // attention block
  ln_kernel<<<ROWS, 128, 0, stream>>>(x, ln1_w, ln1_b, hbuf);
  gemm_kernel<3><<<dim3(3 * DIMSZ / 128, ROWS / 128), 256, 0, stream>>>(
      hbuf, qkvWT, qkv_b, nullptr, qkv16, vT, ROWS, 3 * DIMSZ, DIMSZ);
  attn_kernel<<<dim3(LSEQ / 64, NHEAD, BATCH), 128, 0, stream>>>(qkv16, vT, obuf);
  gemm_kernel<2><<<dim3(DIMSZ / 128, ROWS / 128), 256, 0, stream>>>(
      obuf, outWT, out_b, x, x1, nullptr, ROWS, DIMSZ, DIMSZ);

  // MLP block
  ln_kernel<<<ROWS, 128, 0, stream>>>(x1, ln2_w, ln2_b, h2);
  gemm_kernel<1><<<dim3(MLPD / 128, ROWS / 128), 256, 0, stream>>>(
      h2, fc1WT, fc1_b, nullptr, a16, nullptr, ROWS, MLPD, DIMSZ);
  gemm_kernel<2><<<dim3(DIMSZ / 128, ROWS / 128), 256, 0, stream>>>(
      a16, fc2WT, fc2_b, x1, d_out, nullptr, ROWS, DIMSZ, MLPD);
}
